// CausalSelfAttention_47416438948261
// MI455X (gfx1250) — compile-verified, matrix operand order checked by reading, not run on hardware
//
#include <hip/hip_runtime.h>
#include <hip/hip_bf16.h>

// ---------------- types ----------------
typedef __bf16 v16bf __attribute__((ext_vector_type(16)));
typedef __bf16 v8bf  __attribute__((ext_vector_type(8)));
typedef __bf16 v4bf  __attribute__((ext_vector_type(4)));
typedef float  v8f   __attribute__((ext_vector_type(8)));
typedef int    v4i   __attribute__((ext_vector_type(4)));

union Frag16 { v16bf v; v8bf h[2]; };

#define WMMA_BF16(A, B, C) \
    __builtin_amdgcn_wmma_f32_16x16x32_bf16(false, (A), false, (B), 0, (C), false, false)

// Problem constants
static constexpr int Bn = 4, Tn = 2048, Cn = 1024, Hn = 16, Dn = 64;
static constexpr int Mn = Bn * Tn;                         // 8192 rows
static constexpr size_t BHTD = (size_t)Bn * Hn * Tn * Dn;  // 8388608

// ---------------- CDNA5 async copy global -> LDS (ASYNCcnt-tracked) ----------------
__device__ __forceinline__ void async_copy_b128(__bf16* lds, const __bf16* g) {
#if __has_builtin(__builtin_amdgcn_global_load_async_to_lds_b128)
    __builtin_amdgcn_global_load_async_to_lds_b128(
        (__attribute__((address_space(1))) v4i*)(v4i*)(void*)g,
        (__attribute__((address_space(3))) v4i*)(v4i*)(void*)lds, 0, 0);
#else
    asm volatile("global_load_async_to_lds_b128 %0, %1, off"
                 :: "v"((unsigned)(unsigned long long)lds), "v"(g) : "memory");
#endif
}

template <int N>
__device__ __forceinline__ void wait_asynccnt() {
#if __has_builtin(__builtin_amdgcn_s_wait_asynccnt)
    __builtin_amdgcn_s_wait_asynccnt(N);
#else
    asm volatile("s_wait_asynccnt %0" :: "n"(N) : "memory");
#endif
}

// ---------------- f32 -> bf16 convert ----------------
__global__ void __launch_bounds__(256) f32_to_bf16_kernel(const float* __restrict__ s,
                                                          __bf16* __restrict__ d, int n4) {
    int i = blockIdx.x * blockDim.x + threadIdx.x;
    if (i < n4) {
        float4 v = ((const float4*)s)[i];
        v4bf o;
        o[0] = (__bf16)v.x; o[1] = (__bf16)v.y; o[2] = (__bf16)v.z; o[3] = (__bf16)v.w;
        ((v4bf*)d)[i] = o;
    }
}

// f32 (K x N) -> bf16 transposed (N x K): makes GEMM B-tiles straight row copies.
__global__ void __launch_bounds__(256) f32_to_bf16_T_kernel(const float* __restrict__ s,
                                                            __bf16* __restrict__ d,
                                                            int K, int N) {
    int idx = blockIdx.x * blockDim.x + threadIdx.x;
    if (idx < K * N) {
        int k = idx / N, n = idx - k * N;
        d[(size_t)n * K + k] = (__bf16)s[idx];
    }
}

// ---------------- shared 128x128 WMMA GEMM mainloop (double-buffered async) ----------------
// A:  row-major M x K (ld = K).  WT: row-major N x K (transposed weights, ld = K).
// Block = 256 threads = 8 waves: 2 waves over M (64 rows), 4 waves over N (32 cols).
__device__ __forceinline__ void gemm128x128(const __bf16* __restrict__ A,
                                            const __bf16* __restrict__ WT,
                                            int K, int tileM, int tileN,
                                            __bf16 (*Abuf)[128][32], __bf16 (*Bbuf)[128][32],
                                            v8f acc[4][2]) {
    const int tid  = threadIdx.x;
    const int wave = tid >> 5, lane = tid & 31;
    const int ln = lane & 15, hf = lane >> 4;
    const int wm = wave >> 2, wn = wave & 3;

    // 4 async b128 issues per thread per tile (2 for A, 2 for B)
    auto issue = [&](int kk, int buf) {
        #pragma unroll
        for (int i = 0; i < 2; ++i) {
            int c = tid + i * 256;                 // 512 chunks of 8 elems each
            int row = c >> 2, coff = (c & 3) * 8;
            async_copy_b128(&Abuf[buf][row][coff],
                            A  + (size_t)(tileM + row) * K + kk + coff);
            async_copy_b128(&Bbuf[buf][row][coff],
                            WT + (size_t)(tileN + row) * K + kk + coff);
        }
    };

    const int steps = K >> 5;
    issue(0, 0);
    for (int s = 0; s < steps; ++s) {
        const int buf = s & 1;
        if (s + 1 < steps) { issue((s + 1) << 5, buf ^ 1); wait_asynccnt<4>(); }
        else               { wait_asynccnt<0>(); }
        __syncthreads();                           // tile s visible to all waves

        Frag16 af[4], bfr[2];
        #pragma unroll
        for (int i = 0; i < 4; ++i) {
            int r = wm * 64 + i * 16 + ln;
            af[i].h[0] = *(const v8bf*)&Abuf[buf][r][8 * hf];
            af[i].h[1] = *(const v8bf*)&Abuf[buf][r][16 + 8 * hf];
        }
        #pragma unroll
        for (int j = 0; j < 2; ++j) {
            int n = wn * 32 + j * 16 + ln;
            bfr[j].h[0] = *(const v8bf*)&Bbuf[buf][n][8 * hf];
            bfr[j].h[1] = *(const v8bf*)&Bbuf[buf][n][16 + 8 * hf];
        }
        #pragma unroll
        for (int i = 0; i < 4; ++i)
            #pragma unroll
            for (int j = 0; j < 2; ++j)
                acc[i][j] = WMMA_BF16(af[i].v, bfr[j].v, acc[i][j]);

        __syncthreads();                           // protect buf before it is re-filled
    }
}

// ---------------- QKV projection ----------------
// qkv = x @ w_qkv + b_qkv; scatter Q,K as (B,H,T,D) bf16 and V as (B,H,D,T) bf16.
__global__ void __launch_bounds__(256) qkv_gemm_kernel(const __bf16* __restrict__ X,
                                                       const __bf16* __restrict__ WT,
                                                       const float* __restrict__ bias,
                                                       __bf16* __restrict__ QKV) {
    __shared__ __align__(16) __bf16 Abuf[2][128][32];
    __shared__ __align__(16) __bf16 Bbuf[2][128][32];
    v8f acc[4][2];
    #pragma unroll
    for (int i = 0; i < 4; ++i)
        #pragma unroll
        for (int j = 0; j < 2; ++j)
            #pragma unroll
            for (int e = 0; e < 8; ++e) acc[i][j][e] = 0.f;

    const int tileN = blockIdx.x * 128;   // over 3072
    const int tileM = blockIdx.y * 128;   // over 8192
    gemm128x128(X, WT, Cn, tileM, tileN, Abuf, Bbuf, acc);

    const int tid = threadIdx.x;
    const int wave = tid >> 5, lane = tid & 31;
    const int ln = lane & 15, hf = lane >> 4;
    const int wm = wave >> 2, wn = wave & 3;

    #pragma unroll
    for (int j = 0; j < 2; ++j) {
        const int col = tileN + wn * 32 + j * 16 + ln;   // 0..3071
        const float bv = bias[col];
        const int which = col >> 10;                     // 0=Q,1=K,2=V (uniform per j)
        const int cc = col & 1023;
        const int hh = cc >> 6, dd = cc & 63;
        __bf16* dst = QKV + (size_t)which * BHTD;
        #pragma unroll
        for (int i = 0; i < 4; ++i) {
            #pragma unroll
            for (int e = 0; e < 8; ++e) {
                int m = tileM + wm * 64 + i * 16 + e + 8 * hf;   // 0..8191
                int bb = m >> 11, tt = m & 2047;
                float val = acc[i][j][e] + bv;
                if (which == 2)   // V transposed: (B,H,D,T)
                    dst[((size_t)(bb * Hn + hh) * Dn + dd) * Tn + tt] = (__bf16)val;
                else              // Q,K: (B,H,T,D)
                    dst[((size_t)(bb * Hn + hh) * Tn + tt) * Dn + dd] = (__bf16)val;
            }
        }
    }
}

// ---------------- Flash attention (double-buffered async K/V) ----------------
// Grid: (T/128, B*H). Block: 256 threads = 8 waves; each wave owns 16 query rows.
__global__ void __launch_bounds__(256) attn_kernel(const __bf16* __restrict__ Qg,
                                                   const __bf16* __restrict__ Kg,
                                                   const __bf16* __restrict__ VgT,
                                                   __bf16* __restrict__ Yb) {
    __shared__ __align__(16) __bf16 Ktile[2][32][64];  // keys row-major
    __shared__ __align__(16) __bf16 Vtile[2][64][32];  // V d-major (from (B,H,D,T) slab)
    __shared__ __align__(16) __bf16 Pb[8][16][32];     // per-wave P relayout buffer

    const int tid = threadIdx.x;
    const int wave = tid >> 5, lane = tid & 31;
    const int ln = lane & 15, hf = lane >> 4;
    const int bh = blockIdx.y;                 // b*16 + h
    const int bb = bh >> 4, hh = bh & 15;
    const int bq = blockIdx.x * 128;
    const int qBase = bq + wave * 16;

    const __bf16* Q  = Qg  + (size_t)bh * Tn * Dn;
    const __bf16* K  = Kg  + (size_t)bh * Tn * Dn;
    const __bf16* Vt = VgT + (size_t)bh * Dn * Tn;

    // Q fragments (16 rows x 64 d = 2 k-step frags), loaded once
    Frag16 qf[2];
    {
        const __bf16* qrow = Q + (size_t)(qBase + ln) * Dn;
        #pragma unroll
        for (int s = 0; s < 2; ++s) {
            qf[s].h[0] = *(const v8bf*)(qrow + 32 * s + 8 * hf);
            qf[s].h[1] = *(const v8bf*)(qrow + 32 * s + 16 + 8 * hf);
        }
    }

    v8f o[4];
    float m_run[8], l_run[8];
    #pragma unroll
    for (int e = 0; e < 8; ++e) {
        m_run[e] = -1e30f;
        l_run[e] = 0.f;
        #pragma unroll
        for (int fd = 0; fd < 4; ++fd) o[fd][e] = 0.f;
    }

    // 2 async b128 issues per thread per chunk (1 K, 1 V)
    auto issueKV = [&](int kc, int buf) {
        { int row = tid >> 3, coff = (tid & 7) * 8;     // K: 32x64
          async_copy_b128(&Ktile[buf][row][coff],
                          K + (size_t)(kc * 32 + row) * Dn + coff); }
        { int row = tid >> 2, coff = (tid & 3) * 8;     // V: 64x32
          async_copy_b128(&Vtile[buf][row][coff],
                          Vt + (size_t)row * Tn + kc * 32 + coff); }
    };

    const int nCh = (bq + 128) >> 5;           // causal: only chunks up to this block
    const float scale = 0.125f;                // 1/sqrt(64)

    issueKV(0, 0);
    for (int kc = 0; kc < nCh; ++kc) {
        const int buf = kc & 1;
        if (kc + 1 < nCh) { issueKV(kc + 1, buf ^ 1); wait_asynccnt<2>(); }
        else              { wait_asynccnt<0>(); }
        __syncthreads();

        // S = Q @ K^T for 32 keys (two 16-key sub-blocks)
        float sv[2][8];
        #pragma unroll
        for (int sf = 0; sf < 2; ++sf) {
            v8f sacc = {0.f, 0.f, 0.f, 0.f, 0.f, 0.f, 0.f, 0.f};
            int krow = sf * 16 + ln;
            #pragma unroll
            for (int s = 0; s < 2; ++s) {
                Frag16 kb;
                kb.h[0] = *(const v8bf*)&Ktile[buf][krow][32 * s + 8 * hf];
                kb.h[1] = *(const v8bf*)&Ktile[buf][krow][32 * s + 16 + 8 * hf];
                sacc = WMMA_BF16(qf[s].v, kb.v, sacc);
            }
            int kj = kc * 32 + sf * 16 + ln;
            #pragma unroll
            for (int e = 0; e < 8; ++e) {
                int qi = qBase + e + 8 * hf;
                float val = sacc[e] * scale;
                sv[sf][e] = (kj > qi) ? -1e30f : val;
            }
        }

        // Online softmax (8 rows per lane; 16-lane half-group col reductions)
        __bf16* Pw = &Pb[wave][0][0];
        #pragma unroll
        for (int e = 0; e < 8; ++e) {
            float mx = fmaxf(sv[0][e], sv[1][e]);
            #pragma unroll
            for (int off = 1; off < 16; off <<= 1)
                mx = fmaxf(mx, __shfl_xor(mx, off, 32));
            float mnew = fmaxf(m_run[e], mx);
            float alpha = __expf(m_run[e] - mnew);
            m_run[e] = mnew;
            float p0 = __expf(sv[0][e] - mnew);
            float p1 = __expf(sv[1][e] - mnew);
            float rs = p0 + p1;
            #pragma unroll
            for (int off = 1; off < 16; off <<= 1)
                rs += __shfl_xor(rs, off, 32);
            l_run[e] = l_run[e] * alpha + rs;
            #pragma unroll
            for (int fd = 0; fd < 4; ++fd) o[fd][e] *= alpha;
            Pw[(e + 8 * hf) * 32 + ln]      = (__bf16)p0;   // C-layout -> LDS
            Pw[(e + 8 * hf) * 32 + 16 + ln] = (__bf16)p1;
        }
        asm volatile("s_wait_dscnt 0" ::: "memory");   // wave-private LDS RAW fence

        // O += P @ V
        Frag16 pf;
        pf.h[0] = *(const v8bf*)&Pw[ln * 32 + 8 * hf];
        pf.h[1] = *(const v8bf*)&Pw[ln * 32 + 16 + 8 * hf];
        #pragma unroll
        for (int fd = 0; fd < 4; ++fd) {
            Frag16 vb;
            int dd = fd * 16 + ln;
            vb.h[0] = *(const v8bf*)&Vtile[buf][dd][8 * hf];
            vb.h[1] = *(const v8bf*)&Vtile[buf][dd][16 + 8 * hf];
            o[fd] = WMMA_BF16(pf.v, vb.v, o[fd]);
        }
        __syncthreads();                           // protect buf before re-fill
    }

    // normalize and write y in (B,T,C) layout as bf16
    #pragma unroll
    for (int e = 0; e < 8; ++e) {
        float inv = 1.0f / l_run[e];
        int t = qBase + e + 8 * hf;
        size_t base = ((size_t)bb * Tn + t) * Cn + hh * Dn;
        #pragma unroll
        for (int fd = 0; fd < 4; ++fd)
            Yb[base + fd * 16 + ln] = (__bf16)(o[fd][e] * inv);
    }
}

// ---------------- Output projection: out = y @ w_out + b_out (f32 out) ----------------
__global__ void __launch_bounds__(256) outproj_gemm_kernel(const __bf16* __restrict__ Y,
                                                           const __bf16* __restrict__ WT,
                                                           const float* __restrict__ bias,
                                                           float* __restrict__ Out) {
    __shared__ __align__(16) __bf16 Abuf[2][128][32];
    __shared__ __align__(16) __bf16 Bbuf[2][128][32];
    v8f acc[4][2];
    #pragma unroll
    for (int i = 0; i < 4; ++i)
        #pragma unroll
        for (int j = 0; j < 2; ++j)
            #pragma unroll
            for (int e = 0; e < 8; ++e) acc[i][j][e] = 0.f;

    const int tileN = blockIdx.x * 128;   // over 1024
    const int tileM = blockIdx.y * 128;   // over 8192
    gemm128x128(Y, WT, Cn, tileM, tileN, Abuf, Bbuf, acc);

    const int tid = threadIdx.x;
    const int wave = tid >> 5, lane = tid & 31;
    const int ln = lane & 15, hf = lane >> 4;
    const int wm = wave >> 2, wn = wave & 3;

    #pragma unroll
    for (int j = 0; j < 2; ++j) {
        const int col = tileN + wn * 32 + j * 16 + ln;
        const float bv = bias[col];
        #pragma unroll
        for (int i = 0; i < 4; ++i) {
            #pragma unroll
            for (int e = 0; e < 8; ++e) {
                int m = tileM + wm * 64 + i * 16 + e + 8 * hf;
                Out[(size_t)m * Cn + col] = acc[i][j][e] + bv;
            }
        }
    }
}

// ---------------- host launcher ----------------
extern "C" void kernel_launch(void* const* d_in, const int* in_sizes, int n_in,
                              void* d_out, int out_size, void* d_ws, size_t ws_size,
                              hipStream_t stream) {
    const float* x      = (const float*)d_in[0];   // (B,T,C)
    const float* w_qkv  = (const float*)d_in[1];   // (C,3C)
    const float* b_qkv  = (const float*)d_in[2];   // (3C)
    const float* w_out  = (const float*)d_in[3];   // (C,C)
    const float* b_out  = (const float*)d_in[4];   // (C)
    float* out = (float*)d_out;

    // workspace layout (bf16 elements), total ~88 MB
    __bf16* xb     = (__bf16*)d_ws;                     // M*C        = 8388608
    __bf16* wqkvT  = xb    + (size_t)Mn * Cn;           // 3C*C       = 3145728 (transposed)
    __bf16* woutT  = wqkvT + (size_t)3 * Cn * Cn;       // C*C        = 1048576 (transposed)
    __bf16* qkvb   = woutT + (size_t)Cn * Cn;           // 3 * BHTD   = 25165824
    __bf16* yb     = qkvb  + 3 * BHTD;                  // M*C        = 8388608

    // 1) converts (weights transposed so every GEMM B-tile is a straight async row copy)
    {
        int n4 = Mn * Cn / 4;
        f32_to_bf16_kernel<<<n4 / 256, 256, 0, stream>>>(x, xb, n4);
        int n = Cn * 3 * Cn;
        f32_to_bf16_T_kernel<<<(n + 255) / 256, 256, 0, stream>>>(w_qkv, wqkvT, Cn, 3 * Cn);
        n = Cn * Cn;
        f32_to_bf16_T_kernel<<<(n + 255) / 256, 256, 0, stream>>>(w_out, woutT, Cn, Cn);
    }
    // 2) QKV projection -> Q,K (B,H,T,D), V (B,H,D,T), bf16
    qkv_gemm_kernel<<<dim3(3 * Cn / 128, Mn / 128), 256, 0, stream>>>(xb, wqkvT, b_qkv, qkvb);
    // 3) flash attention -> y (B,T,C) bf16
    attn_kernel<<<dim3(Tn / 128, Bn * Hn), 256, 0, stream>>>(
        qkvb, qkvb + BHTD, qkvb + 2 * BHTD, yb);
    // 4) output projection -> f32 out
    outproj_gemm_kernel<<<dim3(Cn / 128, Mn / 128), 256, 0, stream>>>(yb, woutT, b_out, out);
}
